// CausalAttentionBlock_80590766342870
// MI455X (gfx1250) — compile-verified, matrix operand order checked by reading, not run on hardware
//
#include <hip/hip_runtime.h>
#include <hip/hip_bf16.h>

typedef __attribute__((ext_vector_type(16))) __bf16 v16bf;
typedef __attribute__((ext_vector_type(8)))  __bf16 v8bf;
typedef __attribute__((ext_vector_type(8)))  float  v8f;

#define B_   2
#define S_   2048
#define H_   1024
#define NH_  16
#define DH_  64
#define M_   (B_*S_)   // 4096 rows
#define N3_  (3*H_)    // 3072 qkv cols

#define WMMA_BF16(a, b, c) \
  __builtin_amdgcn_wmma_f32_16x16x32_bf16(false, (a), false, (b), (short)0, (c), false, false)

// ---------------- conversion kernels ----------------

__global__ void k_cvt_bf16(const float* __restrict__ in, __bf16* __restrict__ out, int n) {
  int i = blockIdx.x * blockDim.x + threadIdx.x;
  if (i < n) out[i] = (__bf16)in[i];
}

// out[c*rows + r] = (bf16) in[r*cols + c]   (store W^T so B-fragments read contiguous)
__global__ void k_cvt_transpose(const float* __restrict__ in, __bf16* __restrict__ out,
                                int rows, int cols) {
  int i = blockIdx.x * blockDim.x + threadIdx.x;
  int n = rows * cols;
  if (i < n) {
    int r = i / cols;
    int c = i - r * cols;
    out[(size_t)c * rows + r] = (__bf16)in[i];
  }
}

// ---------------- WMMA fragment helper ----------------
// 16-bit A/B fragment: lane&15 selects row (A) / col (B); lanes 0-15 hold
// K = 0..7 and 16..23, lanes 16-31 hold K = 8..15 and 24..31.
// rowbase points at the 32 contiguous K-values for this lane's row/col.
__device__ __forceinline__ v16bf load_frag_row(const __bf16* rowbase, int lane) {
  int kb = (lane >> 4) * 8;
  v8bf lo = *reinterpret_cast<const v8bf*>(rowbase + kb);
  v8bf hi = *reinterpret_cast<const v8bf*>(rowbase + 16 + kb);
  return __builtin_shufflevector(lo, hi, 0,1,2,3,4,5,6,7,8,9,10,11,12,13,14,15);
}

// async copy of 16 bytes global -> LDS (per lane), tracked by ASYNCcnt.
// VDST = LDS byte address (low 32 bits of flat shared pointer, per aperture rule),
// VADDR = 64-bit global address, GV mode (saddr = off).
__device__ __forceinline__ void async_cp16(const __bf16* gsrc, __bf16* lds_dst) {
  unsigned la = (unsigned)(uintptr_t)lds_dst;
  asm volatile("global_load_async_to_lds_b128 %0, %1, off"
               :: "v"(la), "v"(gsrc) : "memory");
}
__device__ __forceinline__ void async_wait0() {
  asm volatile("s_wait_asynccnt 0" ::: "memory");
}

// ---------------- QKV projection GEMM ----------------
// C(4096 x 3072) = Xb(4096 x 1024) @ Wqkv ; Wqkv supplied transposed (3072 x 1024).
// One wave computes a 16(M) x 64(N) tile (compiler pipelines the simple loop).
// Epilogue scatters into per-head Q (B,NH,S,DH), K (B,NH,S,DH), V^T (B,NH,DH,S).
__global__ void __launch_bounds__(128)
k_gemm_qkv(const __bf16* __restrict__ xb, const __bf16* __restrict__ wT,
           __bf16* __restrict__ Qo, __bf16* __restrict__ Ko,
           __bf16* __restrict__ VTo) {
  int lane = threadIdx.x & 31;
  int wave = threadIdx.x >> 5;
  int n0 = (blockIdx.x * 4 + wave) * 64;
  int m0 = blockIdx.y * 16;
  int arow = m0 + (lane & 15);

  v8f acc[4] = {};
  for (int k0 = 0; k0 < H_; k0 += 32) {
    v16bf a = load_frag_row(xb + (size_t)arow * H_ + k0, lane);
#pragma unroll
    for (int nt = 0; nt < 4; ++nt) {
      int col = n0 + nt * 16 + (lane & 15);
      v16bf b = load_frag_row(wT + (size_t)col * H_ + k0, lane);
      acc[nt] = WMMA_BF16(a, b, acc[nt]);
    }
  }

  // C/D layout: VGPR j -> row (lane>=16 ? 8 : 0)+j, col = lane&15
  int ncol = lane & 15;
  int mb   = (lane >> 4) * 8;
#pragma unroll
  for (int nt = 0; nt < 4; ++nt) {
#pragma unroll
    for (int j = 0; j < 8; ++j) {
      int gm = m0 + mb + j;
      int gc = n0 + nt * 16 + ncol;
      int t = gc >> 10;            // 0=Q 1=K 2=V
      int f = gc & 1023;
      int h = f >> 6;
      int d = f & 63;
      int b = gm >> 11;            // batch
      int s = gm & 2047;           // seq pos
      __bf16 v = (__bf16)acc[nt][j];
      if (t == 0)      Qo[(((size_t)(b * NH_ + h)) * S_ + s) * DH_ + d] = v;
      else if (t == 1) Ko[(((size_t)(b * NH_ + h)) * S_ + s) * DH_ + d] = v;
      else             VTo[(((size_t)(b * NH_ + h)) * DH_ + d) * S_ + s] = v;
    }
  }
}

// ---------------- flash attention ----------------
// One block = 4 waves = 4 consecutive 16-query tiles of one (batch,head).
// The block cooperatively stages each 32-key K tile (32x64 bf16) and V^T tile
// (64x32 bf16) into LDS via global_load_async_to_lds_b128, then each wave runs
// online softmax + WMMA out of LDS. Causal skip is wave-uniform (EXEC stays full).
__global__ void __launch_bounds__(128)
k_flash_attn(const __bf16* __restrict__ Q, const __bf16* __restrict__ K,
             const __bf16* __restrict__ VT, __bf16* __restrict__ attn) {
  __shared__ __align__(16) __bf16 lds_k[32 * 64];      // [key][dh]   4 KB
  __shared__ __align__(16) __bf16 lds_v[64 * 32];      // [feat][key] 4 KB
  __shared__ __align__(16) __bf16 lds_p[4][16 * 32];   // per-wave P staging

  int tid  = threadIdx.x;
  int lane = tid & 31;
  int wave = tid >> 5;
  int q0 = (blockIdx.x * 4 + wave) * 16;
  int q0_max = (blockIdx.x * 4 + 3) * 16;
  int bh = blockIdx.y;                  // b*NH + h
  int b = bh >> 4;
  int h = bh & 15;

  const __bf16* Qb = Q  + (size_t)bh * S_ * DH_;
  const __bf16* Kb = K  + (size_t)bh * S_ * DH_;
  const __bf16* Vb = VT + (size_t)bh * DH_ * S_;

  int ncol = lane & 15;
  int mb   = (lane >> 4) * 8;

  // Q fragments (A-matrix 16 x 64 over dh, as two 16x32 frags)
  int qrow = q0 + ncol;
  v16bf qf0 = load_frag_row(Qb + (size_t)qrow * DH_ + 0,  lane);
  v16bf qf1 = load_frag_row(Qb + (size_t)qrow * DH_ + 32, lane);

  v8f o[4] = {};
  float mrow[8], lrow[8];
#pragma unroll
  for (int j = 0; j < 8; ++j) { mrow[j] = -1e30f; lrow[j] = 0.0f; }

  __bf16* pbuf = &lds_p[wave][0];

  int nblk = (q0_max + 16 + 31) >> 5;   // 32-key blocks covering the block's queries
  for (int blk = 0; blk < nblk; ++blk) {
    int kbase = blk * 32;

    __syncthreads();                    // previous iteration's LDS reads done
    // ---- cooperative async staging: K tile is 2048 contiguous bf16 ----
    {
      const __bf16* gk = Kb + (size_t)kbase * DH_;
#pragma unroll
      for (int c = 0; c < 2; ++c) {
        int e = (tid + c * 128) * 8;               // 0..2040, 16B chunks
        async_cp16(gk + e, lds_k + e);
      }
#pragma unroll
      for (int c = 0; c < 2; ++c) {                // V^T rows: 64 feats x 32 keys
        int e = (tid + c * 128) * 8;
        int f  = e >> 5;
        int kk = e & 31;
        async_cp16(Vb + (size_t)f * S_ + kbase + kk, lds_v + e);
      }
    }
    async_wait0();
    __syncthreads();                    // staged tiles visible to all waves

    if (kbase <= q0 + 15) {             // wave-uniform causal guard
      // ---- scores S = Q K^T for 16 q-rows x 32 keys (two 16-col tiles) ----
      v8f s0 = {}, s1 = {};
      {
        const __bf16* kr = lds_k + (size_t)ncol * DH_;
        v16bf b0 = load_frag_row(kr,      lane);
        v16bf b1 = load_frag_row(kr + 32, lane);
        s0 = WMMA_BF16(qf0, b0, s0);
        s0 = WMMA_BF16(qf1, b1, s0);
      }
      {
        const __bf16* kr = lds_k + (size_t)(16 + ncol) * DH_;
        v16bf b0 = load_frag_row(kr,      lane);
        v16bf b1 = load_frag_row(kr + 32, lane);
        s1 = WMMA_BF16(qf0, b0, s1);
        s1 = WMMA_BF16(qf1, b1, s1);
      }

      // ---- mask + scale + online softmax; stage P into LDS (D->A relayout) ----
#pragma unroll
      for (int j = 0; j < 8; ++j) {
        int qg  = q0 + mb + j;
        int kg0 = kbase + ncol;
        int kg1 = kbase + 16 + ncol;
        float x0 = (kg0 <= qg) ? s0[j] * 0.125f : -1e30f;   // 1/sqrt(64)
        float x1 = (kg1 <= qg) ? s1[j] * 0.125f : -1e30f;
        float rm = fmaxf(x0, x1);                            // row max over 16-lane half
        rm = fmaxf(rm, __shfl_xor(rm, 1, 32));
        rm = fmaxf(rm, __shfl_xor(rm, 2, 32));
        rm = fmaxf(rm, __shfl_xor(rm, 4, 32));
        rm = fmaxf(rm, __shfl_xor(rm, 8, 32));
        float mn   = fmaxf(mrow[j], rm);
        float corr = __expf(mrow[j] - mn);
        float p0   = __expf(x0 - mn);
        float p1   = __expf(x1 - mn);
        float rs = p0 + p1;
        rs += __shfl_xor(rs, 1, 32);
        rs += __shfl_xor(rs, 2, 32);
        rs += __shfl_xor(rs, 4, 32);
        rs += __shfl_xor(rs, 8, 32);
        lrow[j] = lrow[j] * corr + rs;
        mrow[j] = mn;
#pragma unroll
        for (int nt = 0; nt < 4; ++nt) o[nt][j] *= corr;
        pbuf[(mb + j) * 32 + ncol]      = (__bf16)p0;
        pbuf[(mb + j) * 32 + 16 + ncol] = (__bf16)p1;
      }
      asm volatile("s_wait_dscnt 0" ::: "memory");

      // ---- P (16x32) as A-fragment, accumulate O += P @ V from LDS ----
      v16bf pf = load_frag_row(pbuf + (size_t)(lane & 15) * 32, lane);
#pragma unroll
      for (int nt = 0; nt < 4; ++nt) {
        v16bf bv = load_frag_row(lds_v + (size_t)(nt * 16 + ncol) * 32, lane);
        o[nt] = WMMA_BF16(pf, bv, o[nt]);
      }
    }
  }

  // ---- normalize and store to attn (B,S,H) bf16 ----
#pragma unroll
  for (int j = 0; j < 8; ++j) {
    float inv = 1.0f / lrow[j];
    size_t grow = (size_t)(b * S_ + q0 + mb + j);
#pragma unroll
    for (int nt = 0; nt < 4; ++nt) {
      attn[grow * H_ + h * DH_ + nt * 16 + ncol] = (__bf16)(o[nt][j] * inv);
    }
  }
}

// ---------------- output projection GEMM ----------------
// out(4096 x 1024, fp32) = attn_bf(4096 x 1024) @ Wout ; Wout supplied transposed.
__global__ void __launch_bounds__(128)
k_gemm_out(const __bf16* __restrict__ ab, const __bf16* __restrict__ wT,
           float* __restrict__ out) {
  int lane = threadIdx.x & 31;
  int wave = threadIdx.x >> 5;
  int n0 = (blockIdx.x * 4 + wave) * 64;
  int m0 = blockIdx.y * 16;
  int arow = m0 + (lane & 15);

  v8f acc[4] = {};
  for (int k0 = 0; k0 < H_; k0 += 32) {
    v16bf a = load_frag_row(ab + (size_t)arow * H_ + k0, lane);
#pragma unroll
    for (int nt = 0; nt < 4; ++nt) {
      int col = n0 + nt * 16 + (lane & 15);
      v16bf b = load_frag_row(wT + (size_t)col * H_ + k0, lane);
      acc[nt] = WMMA_BF16(a, b, acc[nt]);
    }
  }

  int ncol = lane & 15;
  int mb   = (lane >> 4) * 8;
#pragma unroll
  for (int nt = 0; nt < 4; ++nt) {
#pragma unroll
    for (int j = 0; j < 8; ++j) {
      int gm = m0 + mb + j;
      int gc = n0 + nt * 16 + ncol;
      out[(size_t)gm * H_ + gc] = acc[nt][j];
    }
  }
}

// ---------------- launch ----------------
extern "C" void kernel_launch(void* const* d_in, const int* in_sizes, int n_in,
                              void* d_out, int out_size, void* d_ws, size_t ws_size,
                              hipStream_t stream) {
  (void)in_sizes; (void)n_in; (void)out_size; (void)ws_size;
  const float* x     = (const float*)d_in[0];
  const float* w_qkv = (const float*)d_in[1];
  const float* w_out = (const float*)d_in[2];
  float* out = (float*)d_out;

  // workspace layout (bytes)
  char* ws = (char*)d_ws;
  __bf16* xb    = (__bf16*)(ws + 0);          //  8 MB : x in bf16        (4096x1024)
  __bf16* wqkvT = (__bf16*)(ws + 8388608);    //  6 MB : w_qkv^T bf16     (3072x1024)
  __bf16* woutT = (__bf16*)(ws + 14680064);   //  2 MB : w_out^T bf16     (1024x1024)
  __bf16* Qb    = (__bf16*)(ws + 16777216);   //  8 MB : Q  (B,NH,S,DH)
  __bf16* Kb    = (__bf16*)(ws + 25165824);   //  8 MB : K  (B,NH,S,DH)
  __bf16* VTb   = (__bf16*)(ws + 33554432);   //  8 MB : V^T(B,NH,DH,S)
  __bf16* attnb = (__bf16*)(ws + 41943040);   //  8 MB : attn (B,S,H) bf16

  // 1) precision conversion (+ weight transposes for contiguous B-fragments)
  {
    int n = M_ * H_;
    k_cvt_bf16<<<(n + 255) / 256, 256, 0, stream>>>(x, xb, n);
  }
  {
    int n = H_ * N3_;
    k_cvt_transpose<<<(n + 255) / 256, 256, 0, stream>>>(w_qkv, wqkvT, H_, N3_);
  }
  {
    int n = H_ * H_;
    k_cvt_transpose<<<(n + 255) / 256, 256, 0, stream>>>(w_out, woutT, H_, H_);
  }

  // 2) QKV projection: grid = (3072/(4*64), 4096/16), 4 waves/block
  k_gemm_qkv<<<dim3(N3_ / 256, M_ / 16), 128, 0, stream>>>(xb, wqkvT, Qb, Kb, VTb);

  // 3) flash attention: grid = (S/16/4, B*NH), 4 waves/block
  k_flash_attn<<<dim3(S_ / 64, B_ * NH_), 128, 0, stream>>>(Qb, Kb, VTb, attnb);

  // 4) output projection: grid = (1024/(4*64), 4096/16)
  k_gemm_out<<<dim3(H_ / 256, M_ / 16), 128, 0, stream>>>(attnb, woutT, out);
}
